// VoxelResBackBone8x_70970039599799
// MI455X (gfx1250) — matrix operand order repaired and checked by reading, not compiled, not run on hardware
//
#include <hip/hip_runtime.h>
#include <stdint.h>
#include <stddef.h>

// ---------------------------------------------------------------------------
// Types for CDNA5 WMMA (wave32): 16x16x32 bf16 -> f32 accumulate
// ---------------------------------------------------------------------------
typedef __attribute__((ext_vector_type(16))) __bf16 v16bf;
typedef __attribute__((ext_vector_type(8)))  float  v8f;

__device__ __forceinline__ unsigned short f2bf(float f) {
  unsigned int u = __float_as_uint(f);
  unsigned int r = (u + 0x7fffu + ((u >> 16) & 1u)) >> 16;  // RNE
  return (unsigned short)r;
}
__device__ __forceinline__ float bf2f(unsigned short h) {
  return __uint_as_float(((unsigned int)h) << 16);
}

// tap -> (dz,dy,dx) for 3x3x3 (khkw==9) or 3x1x1 (khkw==1) kernels.
// Exact mul-shift reciprocals for tap < 82 (we only ever have taps <= 27).
__device__ __forceinline__ void tap_dzyx(int tap, int khkw,
                                         int& dz, int& dy, int& dx) {
  if (khkw == 9) {
    dz = (tap * 57) >> 9;        // tap / 9
    int r = tap - dz * 9;
    dy = (r * 86) >> 8;          // r / 3
    dx = r - dy * 3;
  } else {
    dz = tap; dy = 0; dx = 0;
  }
}

// ---------------------------------------------------------------------------
// Scatter active voxels into dense bf16 grid + mask
// ---------------------------------------------------------------------------
__global__ void scatter_kernel(const float* __restrict__ feats,
                               const int* __restrict__ coords, int N,
                               unsigned short* __restrict__ dense,
                               unsigned char* __restrict__ mask,
                               int D, int H, int W, int C) {
  int i = blockIdx.x * blockDim.x + threadIdx.x;
  if (i >= N) return;
  int z = coords[i * 4 + 1];
  int y = coords[i * 4 + 2];
  int x = coords[i * 4 + 3];
  size_t v = ((size_t)z * H + y) * W + x;
  mask[v] = 1;
  for (int c = 0; c < C; ++c) dense[v * C + c] = f2bf(feats[i * C + c]);
}

// ---------------------------------------------------------------------------
// Strided "window-any" mask downsample (new active set for SparseConv3d)
// grid: (ceil(Wo/256), Ho, Do)
// ---------------------------------------------------------------------------
__global__ void mask_down_kernel(const unsigned char* __restrict__ mi,
                                 unsigned char* __restrict__ mo,
                                 int Di, int Hi, int Wi,
                                 int Ho, int Wo,
                                 int kd, int kh, int kw,
                                 int sd, int sh, int sw,
                                 int pd, int ph, int pw) {
  int wo = blockIdx.x * blockDim.x + threadIdx.x;
  if (wo >= Wo) return;
  int ho = blockIdx.y;
  int d0 = blockIdx.z;
  int any = 0;
  for (int dz = 0; dz < kd; ++dz) {
    int iz = d0 * sd - pd + dz;
    if ((unsigned)iz >= (unsigned)Di) continue;
    for (int dy = 0; dy < kh; ++dy) {
      int iy = ho * sh - ph + dy;
      if ((unsigned)iy >= (unsigned)Hi) continue;
      for (int dx = 0; dx < kw; ++dx) {
        int ix = wo * sw - pw + dx;
        if ((unsigned)ix >= (unsigned)Wi) continue;
        any |= mi[((size_t)iz * Hi + iy) * Wi + ix];
      }
    }
  }
  mo[((size_t)d0 * Ho + ho) * Wo + wo] = (unsigned char)(any ? 1 : 0);
}

// ---------------------------------------------------------------------------
// Repack f32 weights (kd,kh,kw,ci,co) -> bf16 [co][Kpad], K = tap*Cin + ci,
// zero-padded so B-fragment loads are always in-bounds.
// grid: (ceil(Kpad/256), Cout, 1); Cin is a power of two.
// ---------------------------------------------------------------------------
__global__ void repack_w_kernel(const float* __restrict__ w,
                                unsigned short* __restrict__ wp,
                                int taps, int Cin, int log2Cin,
                                int Cout, int Kpad) {
  int k = blockIdx.x * blockDim.x + threadIdx.x;
  if (k >= Kpad) return;
  int co = blockIdx.y;
  int tap = k >> log2Cin;
  int ci = k & (Cin - 1);
  float val = (tap < taps) ? w[((size_t)tap * Cin + ci) * Cout + co] : 0.f;
  wp[(size_t)co * Kpad + k] = f2bf(val);
}

// ---------------------------------------------------------------------------
// Implicit-GEMM conv3d with v_wmma_f32_16x16x32_bf16.
// One wave computes a 16(spatial along W) x 16(cout) tile.
// grid: (ceil(wtiles*ntiles/4) blocks of 128 threads, Ho, Do) -> no divisions.
// ISA 7.12.2 fragment layouts:
//   A 16x32 bf16 per lane = two runs of 8 consecutive K:
//     run0 -> VGPR 0-3 : K = kb + (hi?8:0) + 0..7
//     run1 -> VGPR 4-7 : K = kb + 16 + (hi?8:0) + 0..7
//   For Cin multiple of 16 a run never crosses a tap => each run is ONE
//   contiguous 16B chunk of the NDHWC activation tensor => global_load_b128.
//   B 32x16 bf16 per lane = two 16B runs of packed weights (same K as A).
//   C/D f32: VGPR v -> M = v + (hi?8:0), N = lane&15.
// Epilogue: y = acc*a + c0 (fused bias+BN), +residual, ReLU, *mask.
// ---------------------------------------------------------------------------
__global__ void __launch_bounds__(128)
conv_wmma_kernel(const unsigned short* __restrict__ in,
                 const unsigned short* __restrict__ wp,
                 const unsigned char*  __restrict__ mask,
                 const unsigned short* __restrict__ resid,
                 const float* __restrict__ gamma, const float* __restrict__ beta,
                 const float* __restrict__ mean,  const float* __restrict__ var,
                 const float* __restrict__ bias,
                 unsigned short* __restrict__ out_bf,
                 float* __restrict__ out_f32,
                 int Di, int Hi, int Wi, int Cin, int log2Cin,
                 int Ho, int Wo, int Cout, int log2ntiles, int totalX,
                 int khkw, int sd, int sh, int sw,
                 int pd, int ph, int pw,
                 int taps, int Kpad, int do_relu) {
  const int x = blockIdx.x * (blockDim.x >> 5) + (threadIdx.x >> 5);
  if (x >= totalX) return;       // wave-uniform: EXEC stays all-1s for WMMA
  const int nt = x & ((1 << log2ntiles) - 1);
  const int w0t = (x >> log2ntiles) << 4;
  const int ho = blockIdx.y;
  const int d0 = blockIdx.z;

  const int lane = threadIdx.x & 31;
  const int nlo = lane & 15;     // A: M row; B/D: N column
  const int hi = lane >> 4;
  const int hioff = hi << 3;

  const int iz0 = d0 * sd - pd;
  const int iy0 = ho * sh - ph;
  const int ix0 = (w0t + nlo) * sw - pw;

  // B per-lane base: two 16B runs per K-chunk (K = kb+hioff+0..7, kb+16+hioff+0..7).
  const unsigned short* wbase =
      wp + (size_t)((nt << 4) + nlo) * Kpad + (hi ? 8 : 0);

  v8f acc = {0.f, 0.f, 0.f, 0.f, 0.f, 0.f, 0.f, 0.f};

  if (log2Cin >= 4) {
    // ---------------- fast path: Cin in {16,32,64,128} ----------------
#pragma unroll 2
    for (int kb = 0; kb < Kpad; kb += 32) {
      union { v16bf v; uint4 q[2]; } A, B;
#pragma unroll
      for (int r = 0; r < 2; ++r) {
        int kstart = kb + (r << 4) + hioff;
        int tap = kstart >> log2Cin;
        uint4 dv = {0u, 0u, 0u, 0u};
        if (tap < taps) {
          int ci = kstart & (Cin - 1);   // multiple of 8 -> 16B aligned
          int dz, dy, dx;
          tap_dzyx(tap, khkw, dz, dy, dx);
          int iz = iz0 + dz, iy = iy0 + dy, ix = ix0 + dx;
          if ((unsigned)iz < (unsigned)Di && (unsigned)iy < (unsigned)Hi &&
              (unsigned)ix < (unsigned)Wi) {
            size_t voff = (((size_t)iz * Hi + iy) * Wi + ix) * Cin + ci;
            dv = *(const uint4*)(in + voff);   // global_load_b128
          }
        }
        A.q[r] = dv;
      }
      const unsigned short* wb = wbase + kb;
      B.q[0] = *(const uint4*)(wb);            // global_load_b128
      B.q[1] = *(const uint4*)(wb + 16);       // global_load_b128
      __builtin_prefetch(wb + 32, 0, 3);       // next chunk -> global_prefetch_b8
#if defined(__HIP_DEVICE_COMPILE__)
      acc = __builtin_amdgcn_wmma_f32_16x16x32_bf16(
          false, A.v, false, B.v, (short)0, acc, false, false);
#else
      (void)A; (void)B;
#endif
    }
  } else {
    // ---------------- generic path (Cin=4, first layer only) ----------------
    for (int kb = 0; kb < Kpad; kb += 32) {
      union { v16bf v; unsigned int u[8]; uint4 q[2]; } A, B;
#pragma unroll
      for (int vg = 0; vg < 8; ++vg) {
        int kl = (vg < 4 ? (vg << 1) : (16 + ((vg - 4) << 1))) + hioff;
        int k = kb + kl;
        int tap = k >> log2Cin;
        unsigned int dv = 0u;
        if (tap < taps) {
          int ci = k & (Cin - 1);
          int dz, dy, dx;
          tap_dzyx(tap, khkw, dz, dy, dx);
          int iz = iz0 + dz, iy = iy0 + dy, ix = ix0 + dx;
          if ((unsigned)iz < (unsigned)Di && (unsigned)iy < (unsigned)Hi &&
              (unsigned)ix < (unsigned)Wi) {
            size_t voff = (((size_t)iz * Hi + iy) * Wi + ix) * Cin + ci;
            dv = *(const unsigned int*)(in + voff);
          }
        }
        A.u[vg] = dv;
      }
      const unsigned short* wb = wbase + kb;
      B.q[0] = *(const uint4*)(wb);
      B.q[1] = *(const uint4*)(wb + 16);
#if defined(__HIP_DEVICE_COMPILE__)
      acc = __builtin_amdgcn_wmma_f32_16x16x32_bf16(
          false, A.v, false, B.v, (short)0, acc, false, false);
#else
      (void)A; (void)B;
#endif
    }
  }

  // ---- epilogue: fused bias+BN, residual, ReLU, mask ----
  const int co = (nt << 4) + nlo;
  float aa = gamma[co] * rsqrtf(var[co] + 1e-3f);
  float c0 = beta[co] - mean[co] * aa;
  if (bias) c0 += bias[co] * aa;
#pragma unroll
  for (int v = 0; v < 8; ++v) {
    int m = v + hioff;
    int wo = w0t + m;
    if (wo >= Wo) continue;
    size_t vo = ((size_t)d0 * Ho + ho) * Wo + wo;
    float y = acc[v] * aa + c0;
    if (resid) y += bf2f(resid[vo * Cout + co]);
    if (do_relu) y = fmaxf(y, 0.f);
    y = mask[vo] ? y : 0.f;
    if (out_f32) out_f32[vo * Cout + co] = y;
    else         out_bf[vo * Cout + co] = f2bf(y);
  }
}

// ---------------------------------------------------------------------------
// Host orchestration
// ---------------------------------------------------------------------------
extern "C" void kernel_launch(void* const* d_in, const int* in_sizes, int n_in,
                              void* d_out, int out_size, void* d_ws, size_t ws_size,
                              hipStream_t stream) {
  (void)n_in; (void)out_size; (void)ws_size;
  const float* feats = (const float*)d_in[0];
  const int* coords = (const int*)d_in[1];
  const int Nvox = in_sizes[0] / 4;

  auto F = [&](int i) -> const float* { return (const float*)d_in[i]; };

  // Stage dims (match reference conv arithmetic)
  const int D0 = 41, H0 = 200, W0 = 176;
  const int D1 = 21, H1 = 100, W1 = 88;
  const int D2 = 11, H2 = 50,  W2 = 44;
  const int D3 = 5,  H3 = 25,  W3 = 22;
  const int DF = 2,  HF = 25,  WF = 22;
  const size_t V0 = (size_t)D0 * H0 * W0;
  const size_t V1 = (size_t)D1 * H1 * W1;
  const size_t V2 = (size_t)D2 * H2 * W2;
  const size_t V3 = (size_t)D3 * H3 * W3;
  const size_t VF = (size_t)DF * HF * WF;

  // Carve workspace
  char* p = (char*)d_ws;
  auto take = [&](size_t b) -> char* {
    char* r = p; p += (b + 255) & ~(size_t)255; return r;
  };
  unsigned short* dIn = (unsigned short*)take(V0 * 4 * 2);
  unsigned short* A0 = (unsigned short*)take(V0 * 16 * 2);
  unsigned short* T0 = (unsigned short*)take(V0 * 16 * 2);
  unsigned short* A1 = (unsigned short*)take(V1 * 32 * 2);
  unsigned short* T1 = (unsigned short*)take(V1 * 32 * 2);
  unsigned short* A2 = (unsigned short*)take(V2 * 64 * 2);
  unsigned short* T2 = (unsigned short*)take(V2 * 64 * 2);
  unsigned short* A3 = (unsigned short*)take(V3 * 128 * 2);
  unsigned short* T3 = (unsigned short*)take(V3 * 128 * 2);
  unsigned char* m0 = (unsigned char*)take(V0);
  unsigned char* m1 = (unsigned char*)take(V1);
  unsigned char* m2 = (unsigned char*)take(V2);
  unsigned char* m3 = (unsigned char*)take(V3);
  unsigned char* mF = (unsigned char*)take(VF);
  unsigned short* WP = (unsigned short*)take((size_t)3456 * 128 * 2);  // max Kpad*Cout

  // Input scatter
  hipMemsetAsync(dIn, 0, V0 * 4 * 2, stream);
  hipMemsetAsync(m0, 0, V0, stream);
  scatter_kernel<<<(Nvox + 255) / 256, 256, 0, stream>>>(
      feats, coords, Nvox, dIn, m0, D0, H0, W0, 4);

  // Generic conv launcher (repack weights, then WMMA conv)
  auto conv = [&](int wi, int bi, int bni,
                  const unsigned short* in, const unsigned char* mask,
                  const unsigned short* resid,
                  unsigned short* obf, float* of32,
                  int Di, int Hi, int Wi, int Cin,
                  int Do, int Ho, int Wo, int Cout,
                  int kd, int kh, int kw, int sd, int sh, int sw,
                  int pd, int ph, int pw, int relu) {
    int taps = kd * kh * kw;
    int K = taps * Cin;
    int Kpad = (K + 31) & ~31;
    int log2Cin = __builtin_ctz((unsigned)Cin);
    repack_w_kernel<<<dim3((Kpad + 255) / 256, Cout, 1), 256, 0, stream>>>(
        F(wi), WP, taps, Cin, log2Cin, Cout, Kpad);
    int wt = (Wo + 15) / 16, ntl = Cout / 16;
    int log2nt = __builtin_ctz((unsigned)ntl);
    int totalX = wt * ntl;
    conv_wmma_kernel<<<dim3((totalX + 3) / 4, Ho, Do), 128, 0, stream>>>(
        in, WP, mask, resid,
        F(bni), F(bni + 1), F(bni + 2), F(bni + 3),  // gamma, beta, mean, var
        bi >= 0 ? F(bi) : nullptr,
        obf, of32,
        Di, Hi, Wi, Cin, log2Cin,
        Ho, Wo, Cout, log2nt, totalX,
        kh * kw, sd, sh, sw, pd, ph, pw, taps, Kpad, relu);
  };

  auto mdown = [&](const unsigned char* mi, unsigned char* mo,
                   int Di, int Hi, int Wi, int Do, int Ho, int Wo,
                   int kd, int kh, int kw, int sd, int sh, int sw,
                   int pd, int ph, int pw) {
    mask_down_kernel<<<dim3((Wo + 255) / 256, Ho, Do), 256, 0, stream>>>(
        mi, mo, Di, Hi, Wi, Ho, Wo, kd, kh, kw, sd, sh, sw, pd, ph, pw);
  };

  // SparseBasicBlock: conv1->BN->ReLU->conv2->BN->(+res)->ReLU, all masked.
  // base leaves: w1, b1, bn1{g,b,m,v}, w2, b2, bn2{g,b,m,v}
  auto block = [&](int base, unsigned short* act, unsigned short* tmp,
                   const unsigned char* mask, int D, int H, int W, int C) {
    conv(base, base + 1, base + 2, act, mask, nullptr, tmp, nullptr,
         D, H, W, C, D, H, W, C, 3, 3, 3, 1, 1, 1, 1, 1, 1, 1);
    conv(base + 6, base + 7, base + 8, tmp, mask, act, act, nullptr,
         D, H, W, C, D, H, W, C, 3, 3, 3, 1, 1, 1, 1, 1, 1, 1);
  };

  // Param leaf layout in d_in (insertion-order flattening of params dict):
  // 3: conv_input.w  4-7: conv_input.bn{g,b,m,v}
  // 8..19 / 20..31: conv1 blocks
  // 32: conv2.down.w 33-36: bn   37..48 / 49..60: conv2 blocks
  // 61: conv3.down.w 62-65: bn   66..77 / 78..89: conv3 blocks
  // 90: conv4.down.w 91-94: bn   95..106 / 107..118: conv4 blocks
  // 119: conv_out.w  120-123: bn

  // conv_input: SubMConv3d 4->16, k3 p1, no bias
  conv(3, -1, 4, dIn, m0, nullptr, A0, nullptr,
       D0, H0, W0, 4, D0, H0, W0, 16, 3, 3, 3, 1, 1, 1, 1, 1, 1, 1);
  block(8, A0, T0, m0, D0, H0, W0, 16);
  block(20, A0, T0, m0, D0, H0, W0, 16);

  // conv2: SparseConv3d 16->32 s2 p1
  mdown(m0, m1, D0, H0, W0, D1, H1, W1, 3, 3, 3, 2, 2, 2, 1, 1, 1);
  conv(32, -1, 33, A0, m1, nullptr, A1, nullptr,
       D0, H0, W0, 16, D1, H1, W1, 32, 3, 3, 3, 2, 2, 2, 1, 1, 1, 1);
  block(37, A1, T1, m1, D1, H1, W1, 32);
  block(49, A1, T1, m1, D1, H1, W1, 32);

  // conv3: 32->64 s2 p1
  mdown(m1, m2, D1, H1, W1, D2, H2, W2, 3, 3, 3, 2, 2, 2, 1, 1, 1);
  conv(61, -1, 62, A1, m2, nullptr, A2, nullptr,
       D1, H1, W1, 32, D2, H2, W2, 64, 3, 3, 3, 2, 2, 2, 1, 1, 1, 1);
  block(66, A2, T2, m2, D2, H2, W2, 64);
  block(78, A2, T2, m2, D2, H2, W2, 64);

  // conv4: 64->128 s2 pad (0,1,1)
  mdown(m2, m3, D2, H2, W2, D3, H3, W3, 3, 3, 3, 2, 2, 2, 0, 1, 1);
  conv(90, -1, 91, A2, m3, nullptr, A3, nullptr,
       D2, H2, W2, 64, D3, H3, W3, 128, 3, 3, 3, 2, 2, 2, 0, 1, 1, 1);
  block(95, A3, T3, m3, D3, H3, W3, 128);
  block(107, A3, T3, m3, D3, H3, W3, 128);

  // conv_out: 128->128, k=(3,1,1), s=(2,1,1), p=0 -> f32 output
  mdown(m3, mF, D3, H3, W3, DF, HF, WF, 3, 1, 1, 2, 1, 1, 0, 0, 0);
  conv(119, -1, 120, A3, mF, nullptr, nullptr, (float*)d_out,
       D3, H3, W3, 128, DF, HF, WF, 128, 3, 1, 1, 2, 1, 1, 0, 0, 0, 1);
}
